// Seq2Seq_33578054320282
// MI455X (gfx1250) — compile-verified, matrix operand order checked by reading, not run on hardware
//
#include <hip/hip_runtime.h>
#include <hip/hip_bf16.h>

// Problem constants (from reference): V=30001, E=300, H=512, B=64, L=48
#define B_   64
#define L_   48
#define E_   300
#define EP_  320          // E padded to multiple of 32 (WMMA K granularity)
#define H_   512
#define G4H  2048         // 4*H
#define V_   30001
#define VP_  30016        // V padded to multiple of 16 (N tiles)
#define MB_  (L_ * B_)    // 3072 rows, time-major [L,B]
#define TN_  4            // N-tiles per wave in the generic GEMM (16x64 strip)

typedef __attribute__((ext_vector_type(16))) __bf16       v16bf;
typedef __attribute__((ext_vector_type(8)))  float        v8f;
typedef __attribute__((ext_vector_type(4)))  unsigned int v4u;

union FragU { v4u q[2]; v16bf v; };

__device__ __forceinline__ unsigned short f2bf(float x) {
  union { float f; unsigned int u; } c; c.f = x;
  unsigned int r = c.u + 0x7FFFu + ((c.u >> 16) & 1u);   // round-to-nearest-even
  return (unsigned short)(r >> 16);
}

// A-fragment (16x32 bf16) / B-fragment (32x16 bf16, weight stored [N,K] row-major)
// per CDNA5 ISA layout: lane l -> row/col = l&15, khalf = (l>>4)*8;
// VGPR0..3 = K[khalf..khalf+7], VGPR4..7 = K[16+khalf..16+khalf+7].
__device__ __forceinline__ v16bf load_frag32(const unsigned short* __restrict__ row_ptr,
                                             int k0, int lane) {
  FragU f;
  const int khalf = (lane >> 4) << 3;
  f.q[0] = *(const v4u*)(row_ptr + k0 + khalf);
  f.q[1] = *(const v4u*)(row_ptr + k0 + 16 + khalf);
  return f.v;
}

__device__ __forceinline__ v8f wmma_bf16(v16bf a, v16bf b, v8f c) {
  return __builtin_amdgcn_wmma_f32_16x16x32_bf16(false, a, false, b, (short)0, c,
                                                 false, false);
}

__device__ __forceinline__ float sigm(float x) { return 1.0f / (1.0f + expf(-x)); }

// ---------------------------------------------------------------------------
// Generic bf16 GEMM: C[M,N](f32) = A[M,Kpad] * B[N,Kpad]^T + bias, opt tanh.
// One wave per 16x64 strip (TN_=4 tiles): 4 independent WMMA accumulator
// chains for XDL overlap, A fragment reused 4x. 8 waves / block along N.
// ---------------------------------------------------------------------------
__global__ __launch_bounds__(256) void gemm_bf16_kernel(
    const unsigned short* __restrict__ A, const unsigned short* __restrict__ Bm,
    const float* __restrict__ bias, float* __restrict__ C,
    int Ntiles, int Nlog, int K, int lda, int ldb, int ldc, int act)
{
  const int lane   = threadIdx.x & 31;
  const int wv     = threadIdx.x >> 5;
  const int ntBase = (blockIdx.x * 8 + wv) * TN_;
  const int mt     = blockIdx.y;
  if (ntBase >= Ntiles) return;                     // uniform per wave: EXEC stays full
  const int m0 = mt << 4;
  const unsigned short* arow = A + (size_t)(m0 + (lane & 15)) * lda;
  const unsigned short* brow[TN_];
#pragma unroll
  for (int i = 0; i < TN_; ++i) {
    int nt = (ntBase + i < Ntiles) ? (ntBase + i) : (Ntiles - 1);  // clamp (unused when %4==0)
    brow[i] = Bm + (size_t)((nt << 4) + (lane & 15)) * ldb;
  }
  v8f acc[TN_];
#pragma unroll
  for (int i = 0; i < TN_; ++i)
#pragma unroll
    for (int r = 0; r < 8; ++r) acc[i][r] = 0.0f;

  for (int k0 = 0; k0 < K; k0 += 32) {
    __builtin_prefetch(arow + k0 + 128, 0, 1);      // global_prefetch_b8
    v16bf a = load_frag32(arow, k0, lane);          // reused across 4 N-tiles
#pragma unroll
    for (int i = 0; i < TN_; ++i) {
      v16bf b = load_frag32(brow[i], k0, lane);
      acc[i] = wmma_bf16(a, b, acc[i]);             // 4 independent chains
    }
  }

  const int mb = m0 + ((lane >> 4) << 3);
#pragma unroll
  for (int i = 0; i < TN_; ++i) {
    if (ntBase + i >= Ntiles) continue;
    const int n = ((ntBase + i) << 4) + (lane & 15);
    if (n >= Nlog) continue;                        // mask ragged N (V=30001)
    const float bv = bias ? bias[n] : 0.0f;
#pragma unroll
    for (int r = 0; r < 8; ++r) {
      float v = acc[i][r] + bv;
      if (act) v = tanhf(v);
      C[(size_t)(mb + r) * ldc + n] = v;
    }
  }
}

// ---------------------------------------------------------------------------
// Fused LSTM recurrent step: one wave owns a 16x16 tile of H and computes all
// four gate tiles (i,f,g,o) so the nonlinearity runs on in-register tiles.
// acc_j = Gx[t] (precomputed x@Wih^T + b) + h_prev @ Whh^T  (gate j block).
// ---------------------------------------------------------------------------
__global__ __launch_bounds__(32) void lstm_step_kernel(
    const unsigned short* __restrict__ hprev, const float* __restrict__ cprev,
    const unsigned short* __restrict__ Whh,   const float* __restrict__ Gx,
    float* __restrict__ cnew, unsigned short* __restrict__ hnew,
    float* __restrict__ hnew_f, unsigned short* __restrict__ Hall)
{
  const int lane = threadIdx.x & 31;
  const int n0 = blockIdx.x << 4;                   // tile within H (0..511)
  const int m0 = blockIdx.y << 4;                   // tile within B (0..63)
  const unsigned short* arow = hprev + (size_t)(m0 + (lane & 15)) * H_;
  const int nr = n0 + (lane & 15);
  const int n  = nr;
  const int mb = m0 + ((lane >> 4) << 3);
  v8f acc[4];
#pragma unroll
  for (int j = 0; j < 4; ++j)
#pragma unroll
    for (int r = 0; r < 8; ++r)
      acc[j][r] = Gx[(size_t)(mb + r) * G4H + j * H_ + n];
  for (int k0 = 0; k0 < H_; k0 += 32) {
    v16bf a = load_frag32(arow, k0, lane);          // reused across all 4 gates
#pragma unroll
    for (int j = 0; j < 4; ++j) {
      v16bf b = load_frag32(Whh + (size_t)(j * H_ + nr) * H_, k0, lane);
      acc[j] = wmma_bf16(a, b, acc[j]);
    }
  }
#pragma unroll
  for (int r = 0; r < 8; ++r) {
    const size_t off = (size_t)(mb + r) * H_ + n;
    float ig = sigm(acc[0][r]);
    float fg = sigm(acc[1][r]);
    float gg = tanhf(acc[2][r]);
    float og = sigm(acc[3][r]);
    float c2 = fg * cprev[off] + ig * gg;
    float h2 = og * tanhf(c2);
    cnew[off]   = c2;
    hnew_f[off] = h2;
    unsigned short hb = f2bf(h2);
    hnew[off] = hb;
    Hall[off] = hb;
  }
}

// ---------------------------------------------------------------------------
// Small helper kernels
// ---------------------------------------------------------------------------
__global__ void convert_pad_kernel(const float* __restrict__ src,
                                   unsigned short* __restrict__ dst,
                                   int rowsPad, int rowsLog, int cols, int colsPad)
{
  size_t idx = (size_t)blockIdx.x * blockDim.x + threadIdx.x;
  size_t total = (size_t)rowsPad * colsPad;
  if (idx >= total) return;
  int r = (int)(idx / colsPad);
  int c = (int)(idx % colsPad);
  float v = (r < rowsLog && c < cols) ? src[(size_t)r * cols + c] : 0.0f;
  dst[idx] = f2bf(v);
}

// shift=0: encoder inputs emb[src[t]]; shift=1: decoder (t==0 -> 1/100 start vec)
__global__ void gather_embed_kernel(const int* __restrict__ tok,
                                    const float* __restrict__ emb,
                                    unsigned short* __restrict__ X, int shift)
{
  size_t idx = (size_t)blockIdx.x * blockDim.x + threadIdx.x;
  if (idx >= (size_t)MB_ * EP_) return;
  int c = (int)(idx % EP_);
  int m = (int)(idx / EP_);
  int t = m / B_, b = m % B_;
  float v;
  if (shift && t == 0) {
    v = (c < E_) ? 0.01f : 0.0f;
  } else {
    int id = tok[b * L_ + (t - shift)];             // src[t][b] = target_qries[b][t]
    v = (c < E_) ? emb[(size_t)id * E_ + c] : 0.0f;
  }
  X[idx] = f2bf(v);
}

__global__ void zero_state_kernel(unsigned short* __restrict__ hbf, float* __restrict__ c)
{
  int i = blockIdx.x * blockDim.x + threadIdx.x;
  if (i >= B_ * H_) return;
  hbf[i] = 0; c[i] = 0.0f;
}

__global__ void copy_f32_kernel(const float* __restrict__ s, float* __restrict__ d, int n)
{
  int i = blockIdx.x * blockDim.x + threadIdx.x;
  if (i < n) d[i] = s[i];
}

// inco row = [h_layer0 | c_layer0 | h_layer1 | c_layer1]
__global__ void build_inco_kernel(const float* __restrict__ h0, const float* __restrict__ c0,
                                  const float* __restrict__ h1, const float* __restrict__ c1,
                                  float* __restrict__ out_f, unsigned short* __restrict__ out_bf)
{
  int i = blockIdx.x * blockDim.x + threadIdx.x;
  if (i >= B_ * G4H) return;
  int b = i / G4H, c = i % G4H;
  float v;
  if      (c <     H_) v = h0[b * H_ + c];
  else if (c < 2 * H_) v = c0[b * H_ + c - H_];
  else if (c < 3 * H_) v = h1[b * H_ + c - 2 * H_];
  else                 v = c1[b * H_ + c - 3 * H_];
  out_f[i]  = v;
  out_bf[i] = f2bf(v);
}

// deco.reshape(B,4,H): dh = blocks 0,1 ; dc = blocks 2,3
__global__ void extract_dec_kernel(const float* __restrict__ deco, int layer,
                                   unsigned short* __restrict__ hbf, float* __restrict__ c)
{
  int i = blockIdx.x * blockDim.x + threadIdx.x;
  if (i >= B_ * H_) return;
  int b = i / H_, j = i % H_;
  hbf[i] = f2bf(deco[(size_t)b * G4H + layer * H_ + j]);
  c[i]   =       deco[(size_t)b * G4H + 2 * H_ + layer * H_ + j];
}

// ---------------------------------------------------------------------------
extern "C" void kernel_launch(void* const* d_in, const int* in_sizes, int n_in,
                              void* d_out, int out_size, void* d_ws, size_t ws_size,
                              hipStream_t stream)
{
  (void)in_sizes; (void)n_in; (void)out_size; (void)ws_size;
  const int*   tok   = (const int*)  d_in[0];
  // d_in[1] = lengths_qry (unused by reference)
  const float* emb   = (const float*)d_in[2];
  const float* eWih0 = (const float*)d_in[3];
  const float* eWhh0 = (const float*)d_in[4];
  const float* eb0   = (const float*)d_in[5];
  const float* eWih1 = (const float*)d_in[6];
  const float* eWhh1 = (const float*)d_in[7];
  const float* eb1   = (const float*)d_in[8];
  const float* ln1W  = (const float*)d_in[9];
  const float* ln1b  = (const float*)d_in[10];
  const float* ln3W  = (const float*)d_in[11];
  const float* ln3b  = (const float*)d_in[12];
  const float* dWih0 = (const float*)d_in[13];
  const float* dWhh0 = (const float*)d_in[14];
  const float* db0   = (const float*)d_in[15];
  const float* dWih1 = (const float*)d_in[16];
  const float* dWhh1 = (const float*)d_in[17];
  const float* db1   = (const float*)d_in[18];
  const float* outW  = (const float*)d_in[19];
  const float* outb  = (const float*)d_in[20];
  float* out = (float*)d_out;

  // Workspace bump allocator (deterministic, no mallocs — graph-capture safe)
  char* p = (char*)d_ws;
  auto alloc = [&](size_t bytes) -> char* {
    char* r = p; p += (bytes + 255) & ~(size_t)255; return r;
  };
  typedef unsigned short bf;
  bf*    Xenc   = (bf*)alloc((size_t)MB_ * EP_ * 2);
  bf*    Xdec   = (bf*)alloc((size_t)MB_ * EP_ * 2);
  bf*    weWih0 = (bf*)alloc((size_t)G4H * EP_ * 2);
  bf*    weWhh0 = (bf*)alloc((size_t)G4H * H_  * 2);
  bf*    weWih1 = (bf*)alloc((size_t)G4H * H_  * 2);
  bf*    weWhh1 = (bf*)alloc((size_t)G4H * H_  * 2);
  bf*    wdWih0 = (bf*)alloc((size_t)G4H * EP_ * 2);
  bf*    wdWhh0 = (bf*)alloc((size_t)G4H * H_  * 2);
  bf*    wdWih1 = (bf*)alloc((size_t)G4H * H_  * 2);
  bf*    wdWhh1 = (bf*)alloc((size_t)G4H * H_  * 2);
  bf*    wln1   = (bf*)alloc((size_t)G4H * G4H * 2);
  bf*    wln3   = (bf*)alloc((size_t)G4H * G4H * 2);
  bf*    wout   = (bf*)alloc((size_t)VP_ * H_  * 2);
  float* Gx     = (float*)alloc((size_t)MB_ * G4H * 4);   // reused per layer
  bf*    Ha     = (bf*)alloc((size_t)MB_ * H_ * 2);
  bf*    Hb     = (bf*)alloc((size_t)MB_ * H_ * 2);
  bf*    hpp[2] = {(bf*)alloc(B_*H_*2), (bf*)alloc(B_*H_*2)};
  float* cpp[2] = {(float*)alloc(B_*H_*4), (float*)alloc(B_*H_*4)};
  float* hf32   = (float*)alloc(B_*H_*4);
  float* h0e    = (float*)alloc(B_*H_*4);
  float* c0e    = (float*)alloc(B_*H_*4);
  float* h1e    = (float*)alloc(B_*H_*4);
  float* c1e    = (float*)alloc(B_*H_*4);
  bf*    incoBf = (bf*)alloc((size_t)B_*G4H*2);
  bf*    neckBf = (bf*)alloc((size_t)B_*G4H*2);

  const size_t offNeck = (size_t)L_ * B_ * V_;            // return order: logits,neck,inco,deco
  const size_t offInco = offNeck + (size_t)B_ * G4H;
  const size_t offDeco = offInco + (size_t)B_ * G4H;

  auto convert = [&](const float* s, bf* d, int rp, int rl, int c, int cp) {
    size_t tot = (size_t)rp * cp;
    convert_pad_kernel<<<(unsigned)((tot + 255) / 256), 256, 0, stream>>>(s, d, rp, rl, c, cp);
  };
  auto gemm = [&](const bf* A, const bf* Bm, const float* bias, float* C,
                  int Mtiles, int Ntiles, int Nlog, int K, int lda, int ldb, int ldc, int act) {
    const unsigned gx = (unsigned)((Ntiles + 8 * TN_ - 1) / (8 * TN_));
    gemm_bf16_kernel<<<dim3(gx, (unsigned)Mtiles), 256, 0, stream>>>(
        A, Bm, bias, C, Ntiles, Nlog, K, lda, ldb, ldc, act);
  };

  // 1) weights -> bf16 (padded where K or N not tile-aligned)
  convert(eWih0, weWih0, G4H, G4H, E_, EP_);
  convert(eWhh0, weWhh0, G4H, G4H, H_, H_);
  convert(eWih1, weWih1, G4H, G4H, H_, H_);
  convert(eWhh1, weWhh1, G4H, G4H, H_, H_);
  convert(dWih0, wdWih0, G4H, G4H, E_, EP_);
  convert(dWhh0, wdWhh0, G4H, G4H, H_, H_);
  convert(dWih1, wdWih1, G4H, G4H, H_, H_);
  convert(dWhh1, wdWhh1, G4H, G4H, H_, H_);
  convert(ln1W,  wln1,   G4H, G4H, G4H, G4H);
  convert(ln3W,  wln3,   G4H, G4H, G4H, G4H);
  convert(outW,  wout,   VP_, V_,  H_,  H_);

  // 2) embedding gathers (time-major [L,B,EP] bf16)
  {
    size_t tot = (size_t)MB_ * EP_;
    gather_embed_kernel<<<(unsigned)((tot + 255) / 256), 256, 0, stream>>>(tok, emb, Xenc, 0);
    gather_embed_kernel<<<(unsigned)((tot + 255) / 256), 256, 0, stream>>>(tok, emb, Xdec, 1);
  }

  // One LSTM layer: big input-proj GEMM (all timesteps), then 48 fused steps.
  auto run_layer = [&](const bf* Xin, int Kin, const bf* Wih, const float* bias,
                       const bf* Whh, bf* Hout, bool zeroInit, const float* decoSrc,
                       int decLayer, float* saveH, float* saveC) {
    gemm(Xin, Wih, bias, Gx, MB_ / 16, G4H / 16, G4H, Kin, Kin, Kin, G4H, 0);
    if (zeroInit)
      zero_state_kernel<<<(B_*H_ + 255) / 256, 256, 0, stream>>>(hpp[0], cpp[0]);
    else
      extract_dec_kernel<<<(B_*H_ + 255) / 256, 256, 0, stream>>>(decoSrc, decLayer,
                                                                  hpp[0], cpp[0]);
    int cur = 0;
    for (int t = 0; t < L_; ++t) {
      lstm_step_kernel<<<dim3(H_ / 16, B_ / 16), 32, 0, stream>>>(
          hpp[cur], cpp[cur], Whh, Gx + (size_t)t * B_ * G4H,
          cpp[cur ^ 1], hpp[cur ^ 1], hf32, Hout + (size_t)t * B_ * H_);
      cur ^= 1;
    }
    if (saveH) {
      copy_f32_kernel<<<(B_*H_ + 255) / 256, 256, 0, stream>>>(hf32, saveH, B_*H_);
      copy_f32_kernel<<<(B_*H_ + 255) / 256, 256, 0, stream>>>(cpp[cur], saveC, B_*H_);
    }
  };

  // 3) encoder (zero initial state)
  run_layer(Xenc, EP_, weWih0, eb0, weWhh0, Ha, true, nullptr, 0, h0e, c0e);
  run_layer(Ha,   H_,  weWih1, eb1, weWhh1, Hb, true, nullptr, 0, h1e, c1e);

  // 4) bottleneck: inco -> tanh(inco@ln1^T+b) -> deco  (neck/inco/deco land in d_out)
  build_inco_kernel<<<(B_*G4H + 255) / 256, 256, 0, stream>>>(h0e, c0e, h1e, c1e,
                                                              out + offInco, incoBf);
  gemm(incoBf, wln1, ln1b, out + offNeck, B_ / 16, G4H / 16, G4H, G4H, G4H, G4H, G4H, 1);
  convert(out + offNeck, neckBf, B_, B_, G4H, G4H);
  gemm(neckBf, wln3, ln3b, out + offDeco, B_ / 16, G4H / 16, G4H, G4H, G4H, G4H, G4H, 0);

  // 5) decoder (initial states from deco blocks)
  run_layer(Xdec, EP_, wdWih0, db0, wdWhh0, Ha, false, out + offDeco, 0, nullptr, nullptr);
  run_layer(Ha,   H_,  wdWih1, db1, wdWhh1, Hb, false, out + offDeco, 1, nullptr, nullptr);

  // 6) output projection: logits[3072, 30001] = d1 @ out_W^T + out_b
  gemm(Hb, wout, outb, out, MB_ / 16, VP_ / 16, V_, H_, H_, H_, V_, 0);
}
